// MixtralStyleMoE_71640054497662
// MI455X (gfx1250) — compile-verified, matrix operand order checked by reading, not run on hardware
//
#include <hip/hip_runtime.h>
#include <hip/hip_bf16.h>

// ---------------- problem constants (from reference) ----------------
constexpr int kH = 1024;   // hidden
constexpr int kF = 2048;   // ffn
constexpr int kE = 8;      // experts
constexpr int kT = 4096;   // tokens = B*S
constexpr int kMT = 32;    // token tile (M) per block (2 sub-tiles of 16)
constexpr int kFC = 128;   // F chunk per iteration (8 column-slices of 16)
constexpr int kKS = 64;    // K-slab for gate/up weight staging
constexpr int kDS = 32;    // K-slab for down weight staging

typedef __bf16        v16bf __attribute__((ext_vector_type(16)));
typedef float         v8f   __attribute__((ext_vector_type(8)));
typedef unsigned int  v4u   __attribute__((ext_vector_type(4)));
typedef int           v8i   __attribute__((ext_vector_type(8)));
typedef int           v4i   __attribute__((ext_vector_type(4)));

#if defined(__has_builtin)
#if __has_builtin(__builtin_amdgcn_tensor_load_to_lds) && \
    __has_builtin(__builtin_amdgcn_s_wait_tensorcnt)
#define USE_TDM 1
#endif
#endif
#ifndef USE_TDM
#define USE_TDM 0
#endif

union Frag {
  v16bf v;
  unsigned u[8];
};

// fp32 -> bf16 RNE
__device__ __forceinline__ unsigned short f2bf(float f) {
  unsigned x = __float_as_uint(f);
  unsigned r = (x + 0x7FFFu + ((x >> 16) & 1u)) >> 16;
  return (unsigned short)r;
}

__device__ __forceinline__ float fast_rcp(float x) {
#if __has_builtin(__builtin_amdgcn_rcpf)
  return __builtin_amdgcn_rcpf(x);
#else
  return 1.0f / x;
#endif
}

// A-fragment from row-major bf16 LDS tile (rowpitch elems).
// ISA 7.12.2 (16-bit A 16x32): lanes 0-15: V0..3 -> K0..7, V4..7 -> K16..23;
//                              lanes 16-31: V0..3 -> K8..15, V4..7 -> K24..31.
// K-pairs are contiguous -> whole fragment is two ds_read_b128 per lane.
__device__ __forceinline__ void load_a_frag_lds(const unsigned short* lds,
                                                int rowpitch, int kbase,
                                                int lane, unsigned out[8]) {
  int g = lane >> 4;
  int row = lane & 15;
  const uint4* p = (const uint4*)(lds + row * rowpitch + kbase);
  uint4 q0 = p[g];       // K = kbase + 8g .. +8g+7
  uint4 q1 = p[2 + g];   // K = kbase + 16 + 8g ..
  out[0] = q0.x; out[1] = q0.y; out[2] = q0.z; out[3] = q0.w;
  out[4] = q1.x; out[5] = q1.y; out[6] = q1.z; out[7] = q1.w;
}

// B-fragment from column-major staged LDS (column owns `pitch` contiguous K).
__device__ __forceinline__ void load_b_frag_lds(const unsigned short* buf,
                                                int pitch, int ncolBase,
                                                int kc, int lane,
                                                unsigned out[8]) {
  int g = lane >> 4;
  int col = ncolBase + (lane & 15);
  const uint4* p = (const uint4*)(buf + col * pitch + kc);
  uint4 q0 = p[g];
  uint4 q1 = p[2 + g];
  out[0] = q0.x; out[1] = q0.y; out[2] = q0.z; out[3] = q0.w;
  out[4] = q1.x; out[5] = q1.y; out[6] = q1.z; out[7] = q1.w;
}

#if USE_TDM
// Issue one 2D TDM tile load: bf16 tile [tileD1 rows][tileD0 contiguous elems]
// from a [tensorD1][tensorD0] tensor (row stride strideD0 elems) into LDS,
// packed linearly (fragment-ready column-major). D# per ISA 08 sect. 8.
__device__ __forceinline__ void tdm_load_tile_2d(unsigned ldsOff,
                                                 const void* gptr,
                                                 unsigned tensorD0,
                                                 unsigned tensorD1,
                                                 unsigned tileD0,
                                                 unsigned tileD1,
                                                 unsigned strideD0) {
  unsigned long long ga = (unsigned long long)(size_t)gptr;
  v4u g0;
  g0[0] = 1u;                                           // count=1, user D#
  g0[1] = ldsOff;                                       // lds_addr
  g0[2] = (unsigned)ga;                                 // global_addr[31:0]
  g0[3] = (unsigned)((ga >> 32) & 0x01FFFFFFu) | (2u << 30);  // [56:32]|type=2
  v8i g1;
  g1[0] = 0x00010000;                                   // data_size=1 (2 bytes)
  g1[1] = (int)((tensorD0 & 0xFFFFu) << 16);            // tensor_dim0[15:0]
  g1[2] = (int)((tensorD0 >> 16) | ((tensorD1 & 0xFFFFu) << 16));
  g1[3] = (int)((tensorD1 >> 16) | (tileD0 << 16));     // tile_dim0
  g1[4] = (int)(tileD1 & 0xFFFFu);                      // tile_dim1 (dim2=0)
  g1[5] = (int)strideD0;                                // tensor_dim0_stride
  g1[6] = 0;
  g1[7] = 0;
  v4i z4 = {0, 0, 0, 0};
  v8i z8 = {0, 0, 0, 0, 0, 0, 0, 0};
  // 6-arg form (clang-23 / therock headers): groups 0..3 + group for 5D + cpol
  __builtin_amdgcn_tensor_load_to_lds(g0, g1, z4, z4, z8, 0);
}
#endif

// Fallback: cooperative column-copy of pre-quantized bf16 weights into LDS.
__device__ __forceinline__ void stage_copy_cols(unsigned short* dst,
                                                const unsigned short* src,
                                                int srcPitch, int ncols,
                                                int kelems, int tid, int nthr) {
  int chunks = ncols * (kelems / 8);
  for (int i = tid; i < chunks; i += nthr) {
    int col = i / (kelems / 8);
    int kk = (i % (kelems / 8)) * 8;
    *(uint4*)(dst + col * kelems + kk) =
        *(const uint4*)(src + (size_t)col * srcPitch + kk);
  }
}

// ---------------- kernel 0: reset per-expert counters ----------------
__global__ void moe_zero_kernel(int* counts) {
  if (threadIdx.x < kE) counts[threadIdx.x] = 0;
}

// ---------------- kernel Qa: hidden fp32 -> bf16 (row-major) ----------------
__global__ __launch_bounds__(256) void quant_hidden_kernel(
    const float* __restrict__ in, unsigned short* __restrict__ out) {
  size_t i = (size_t)blockIdx.x * 256 + threadIdx.x;  // over float4 groups
  float4 v = ((const float4*)in)[i];
  uint2 r;
  r.x = ((unsigned)f2bf(v.y) << 16) | f2bf(v.x);
  r.y = ((unsigned)f2bf(v.w) << 16) | f2bf(v.z);
  ((uint2*)out)[i] = r;
}

// ------- kernel Qb: per-expert transpose+quantize [R][C] f32 -> [C][R] bf16 -
__global__ __launch_bounds__(256) void transq_kernel(
    const float* __restrict__ in, unsigned short* __restrict__ out,
    int R, int C) {
  __shared__ float tile[32][33];
  const float* src = in + (size_t)blockIdx.z * R * C;
  unsigned short* dst = out + (size_t)blockIdx.z * R * C;
  int tx = threadIdx.x, ty = threadIdx.y;  // blockDim (32, 8)
  int r0 = blockIdx.y * 32, c0 = blockIdx.x * 32;
#pragma unroll
  for (int i = 0; i < 4; ++i)
    tile[ty + i * 8][tx] = src[(size_t)(r0 + ty + i * 8) * C + c0 + tx];
  __syncthreads();
#pragma unroll
  for (int i = 0; i < 4; ++i) {
    int oc = ty + i * 8;  // output row = original column
    dst[(size_t)(c0 + oc) * R + r0 + tx] = f2bf(tile[tx][oc]);
  }
}

// ---------------- kernel 1: router (one wave32 per token) ----------------
__global__ __launch_bounds__(256) void moe_router_kernel(
    const float* __restrict__ hidden, const float* __restrict__ gate_w,
    int* __restrict__ counts, int* __restrict__ list,
    float* __restrict__ topw) {
  int lane = threadIdx.x & 31;
  int wave = threadIdx.x >> 5;
  int t = blockIdx.x * 8 + wave;
  if (t >= kT) return;

  float acc[kE];
#pragma unroll
  for (int e = 0; e < kE; ++e) acc[e] = 0.f;

  const float* x = hidden + (size_t)t * kH;
  for (int h = lane; h < kH; h += 32) {
    float xv = x[h];
    const float* gw = gate_w + h * kE;  // [H, E] row-major
#pragma unroll
    for (int e = 0; e < kE; ++e) acc[e] += xv * gw[e];
  }
#pragma unroll
  for (int e = 0; e < kE; ++e) {
#pragma unroll
    for (int off = 16; off > 0; off >>= 1)
      acc[e] += __shfl_xor(acc[e], off, 32);
  }
  if (lane == 0) {
    float m = acc[0];
#pragma unroll
    for (int e = 1; e < kE; ++e) m = fmaxf(m, acc[e]);
    float p[kE], s = 0.f;
#pragma unroll
    for (int e = 0; e < kE; ++e) { p[e] = __expf(acc[e] - m); s += p[e]; }
    float rs = fast_rcp(s);
#pragma unroll
    for (int e = 0; e < kE; ++e) p[e] *= rs;
    int i0 = 0;
#pragma unroll
    for (int e = 1; e < kE; ++e) if (p[e] > p[i0]) i0 = e;
    int i1 = (i0 == 0) ? 1 : 0;
#pragma unroll
    for (int e = 0; e < kE; ++e) if (e != i0 && p[e] > p[i1]) i1 = e;
    float w0 = p[i0], w1 = p[i1];
    float rw = fast_rcp(w0 + w1);
    topw[t * 2 + 0] = w0 * rw;
    topw[t * 2 + 1] = w1 * rw;
    int pos0 = atomicAdd(&counts[i0], 1);
    list[i0 * kT + pos0] = (t << 1) | 0;
    int pos1 = atomicAdd(&counts[i1], 1);
    list[i1 * kT + pos1] = (t << 1) | 1;
  }
}

// ---------------- kernel 2: fused expert SwiGLU GEMM (bf16 WMMA) ------------
// grid = (kT/kMT, kE), block = 512 (16 wave32: 2 M-subtiles x 8 H-slices)
__global__ __launch_bounds__(512) void moe_expert_kernel(
    const unsigned short* __restrict__ hbf,  // [T][H] bf16
    const unsigned short* __restrict__ wqg,  // [E][F][H] bf16 (K=H contiguous)
    const unsigned short* __restrict__ wqu,  // [E][F][H] bf16
    const unsigned short* __restrict__ wqd,  // [E][H][F] bf16 (K=F contiguous)
    const int* __restrict__ counts,
    const int* __restrict__ list,            // [E][kT] packed (t<<1)|k
    float* __restrict__ ybuf) {              // [2*kT][H]
  __shared__ __align__(16) unsigned short ldsX[kMT * kH];     // 64 KB X tile
  __shared__ __align__(16) unsigned short ldsAct[kMT * kFC];  // 8 KB act chunk
  // Shared staging buffer: G/U phase = Wg (16KB) + Wu (16KB); Y phase = Wd 64KB
  __shared__ __align__(16) unsigned short ldsW[kH * kDS];     // 64 KB
  __shared__ int ldsSlot[kMT];

  const int e = blockIdx.y;
  const int cnt = counts[e];
  const int tileBase = blockIdx.x * kMT;
  if (tileBase >= cnt) return;  // block-uniform: EXEC stays all-ones below

  const int tid = threadIdx.x;
  const int lane = tid & 31;
  const int wave = tid >> 5;   // 0..15
  const int hs = wave & 7;     // H-slice / F-column slice
  const int ms = wave >> 3;    // M sub-tile (rows ms*16 .. +15)

  unsigned short* ldsWg = ldsW;                 // [128 cols][64 K]
  unsigned short* ldsWu = ldsW + kFC * kKS;     // [128 cols][64 K]

  if (tid < kMT) {
    int idx = tileBase + tid;
    ldsSlot[tid] = (idx < cnt) ? list[e * kT + idx] : -1;
  }
  __syncthreads();

  // Gather token rows (already bf16) -> LDS X tile. Pure b128 copies.
  {
    int row = tid >> 4;   // 32 rows
    int sub = tid & 15;   // 16 threads per row
    int packed = ldsSlot[row];
    int tok = (packed >= 0) ? (packed >> 1) : 0;
    const uint4* src = (const uint4*)(hbf + (size_t)tok * kH);
    uint4* dst = (uint4*)&ldsX[row * kH];
#pragma unroll
    for (int j = sub; j < kH / 8; j += 16) dst[j] = src[j];
  }

  const unsigned short* Wg = wqg + (size_t)e * kH * kF;
  const unsigned short* Wu = wqu + (size_t)e * kH * kF;
  const unsigned short* Wd = wqd + (size_t)e * kH * kF;

  v8f yAcc[8];
  v8f vzero = {};
#pragma unroll
  for (int t = 0; t < 8; ++t) yAcc[t] = vzero;

  const int fcol = hs * 16;    // wave's G/U/act column slice within chunk
  const int h0 = hs * 128;     // wave's Y (output H) column base
  const unsigned short* ldsXm = ldsX + ms * 16 * kH;
  const unsigned short* ldsActm = ldsAct + ms * 16 * kFC;

  for (int f0 = 0; f0 < kF; f0 += kFC) {
    // ---- G, U tiles: [16 tokens x 16 f] accumulated over K = H ----
    v8f gAcc = vzero, uAcc = vzero;
    for (int k0 = 0; k0 < kH; k0 += kKS) {
      __syncthreads();  // previous staging consumers done
#if USE_TDM
      if (wave == 0) {
        // Wg/Wu tiles: 128 F-columns x 64 contiguous K(H) elems each.
        tdm_load_tile_2d((unsigned)(size_t)ldsWg,
                         Wg + (size_t)f0 * kH + k0, kH, kF, kKS, kFC, kH);
        tdm_load_tile_2d((unsigned)(size_t)ldsWu,
                         Wu + (size_t)f0 * kH + k0, kH, kF, kKS, kFC, kH);
        __builtin_amdgcn_s_wait_tensorcnt(0);
      }
#else
      stage_copy_cols(ldsWg, Wg + (size_t)f0 * kH + k0, kH, kFC, kKS, tid, 512);
      stage_copy_cols(ldsWu, Wu + (size_t)f0 * kH + k0, kH, kFC, kKS, tid, 512);
#endif
      __syncthreads();
#pragma unroll
      for (int kc = 0; kc < kKS; kc += 32) {
        Frag a, bg, bu;
        load_a_frag_lds(ldsXm, kH, k0 + kc, lane, a.u);
        load_b_frag_lds(ldsWg, kKS, fcol, kc, lane, bg.u);
        load_b_frag_lds(ldsWu, kKS, fcol, kc, lane, bu.u);
        gAcc = __builtin_amdgcn_wmma_f32_16x16x32_bf16(
            false, a.v, false, bg.v, (short)0, gAcc, false, false);
        uAcc = __builtin_amdgcn_wmma_f32_16x16x32_bf16(
            false, a.v, false, bu.v, (short)0, uAcc, false, false);
      }
    }

    // ---- act = silu(G) * U -> shared bf16 chunk [32 x 128] ----
    __syncthreads();  // previous iteration's act readers done
    {
      int g_ = lane >> 4;
      int col = fcol + (lane & 15);
#pragma unroll
      for (int i = 0; i < 8; ++i) {
        float gv = gAcc[i], uv = uAcc[i];
        float s = gv * fast_rcp(1.0f + __expf(-gv));  // silu via v_rcp_f32
        ldsAct[(ms * 16 + i + 8 * g_) * kFC + col] = f2bf(s * uv);
      }
    }

    // ---- Y[32 x 128/wave] += act[32 x 128] @ Wd[f0:f0+128, :] ----
    for (int ks = 0; ks < kFC; ks += kDS) {
      __syncthreads();  // act written / previous Wd slab consumers done
#if USE_TDM
      if (wave == 0) {
        // Wd tile: 1024 H-columns x 32 contiguous K(F) elems each.
        tdm_load_tile_2d((unsigned)(size_t)ldsW,
                         Wd + f0 + ks, kF, kH, kDS, kH, kF);
        __builtin_amdgcn_s_wait_tensorcnt(0);
      }
#else
      stage_copy_cols(ldsW, Wd + f0 + ks, kF, kH, kDS, tid, 512);
#endif
      __syncthreads();
      Frag a;
      load_a_frag_lds(ldsActm, kFC, ks, lane, a.u);
#pragma unroll
      for (int t = 0; t < 8; ++t) {
        Frag b;
        load_b_frag_lds(ldsW, kDS, h0 + t * 16, 0, lane, b.u);
        yAcc[t] = __builtin_amdgcn_wmma_f32_16x16x32_bf16(
            false, a.v, false, b.v, (short)0, yAcc[t], false, false);
      }
    }
  }

  // ---- scatter Y rows to per-assignment slots (slot = t*2 + k) ----
  {
    int g_ = lane >> 4;
    int col = lane & 15;
#pragma unroll
    for (int t = 0; t < 8; ++t) {
#pragma unroll
      for (int i = 0; i < 8; ++i) {
        int row = ms * 16 + i + 8 * g_;
        int packed = ldsSlot[row];
        if (packed >= 0)
          ybuf[(size_t)packed * kH + h0 + t * 16 + col] = yAcc[t][i];
      }
    }
  }
}

// ---------------- kernel 3: weighted combine of the two expert outputs ------
__global__ __launch_bounds__(256) void moe_combine_kernel(
    const float* __restrict__ ybuf, const float* __restrict__ topw,
    float* __restrict__ out) {
  int t = blockIdx.x;
  float w0 = topw[t * 2 + 0];
  float w1 = topw[t * 2 + 1];
  const float4* y0 = (const float4*)(ybuf + (size_t)(t * 2 + 0) * kH);
  const float4* y1 = (const float4*)(ybuf + (size_t)(t * 2 + 1) * kH);
  float4* o = (float4*)(out + (size_t)t * kH);
  int i = threadIdx.x;  // kH/4 == 256
  float4 a = y0[i], b = y1[i];
  float4 r;
  r.x = w0 * a.x + w1 * b.x;
  r.y = w0 * a.y + w1 * b.y;
  r.z = w0 * a.z + w1 * b.z;
  r.w = w0 * a.w + w1 * b.w;
  o[i] = r;
}

// ---------------- host-side launch ----------------
extern "C" void kernel_launch(void* const* d_in, const int* in_sizes, int n_in,
                              void* d_out, int out_size, void* d_ws,
                              size_t ws_size, hipStream_t stream) {
  const float* hidden    = (const float*)d_in[0];  // [B,S,H]
  const float* gate_w    = (const float*)d_in[1];  // [H,E]
  const float* gate_proj = (const float*)d_in[2];  // [E,H,F]
  const float* up_proj   = (const float*)d_in[3];  // [E,H,F]
  const float* down_proj = (const float*)d_in[4];  // [E,F,H]
  float* out = (float*)d_out;

  char* ws = (char*)d_ws;
  size_t off = 0;
  int* counts = (int*)(ws + off);              off += 256;
  float* topw = (float*)(ws + off);            off += (size_t)kT * 2 * 4;
  int* list   = (int*)(ws + off);              off += (size_t)kE * kT * 4;
  float* ybuf = (float*)(ws + off);            off += (size_t)2 * kT * kH * 4;
  unsigned short* hbf = (unsigned short*)(ws + off);  off += (size_t)kT * kH * 2;
  unsigned short* wqg = (unsigned short*)(ws + off);  off += (size_t)kE * kH * kF * 2;
  unsigned short* wqu = (unsigned short*)(ws + off);  off += (size_t)kE * kH * kF * 2;
  unsigned short* wqd = (unsigned short*)(ws + off);  off += (size_t)kE * kH * kF * 2;

  moe_zero_kernel<<<1, 32, 0, stream>>>(counts);

  // One-time quantize / transpose into fragment-ready bf16 layouts.
  quant_hidden_kernel<<<(kT * kH / 4) / 256, 256, 0, stream>>>(hidden, hbf);
  dim3 tb(32, 8);
  dim3 tg_gu(kF / 32, kH / 32, kE);   // [H][F] -> [F][H]
  transq_kernel<<<tg_gu, tb, 0, stream>>>(gate_proj, wqg, kH, kF);
  transq_kernel<<<tg_gu, tb, 0, stream>>>(up_proj, wqu, kH, kF);
  dim3 tg_d(kH / 32, kF / 32, kE);    // [F][H] -> [H][F]
  transq_kernel<<<tg_d, tb, 0, stream>>>(down_proj, wqd, kF, kH);

  moe_router_kernel<<<kT / 8, 256, 0, stream>>>(hidden, gate_w, counts, list,
                                                topw);
  dim3 grid(kT / kMT, kE);
  moe_expert_kernel<<<grid, 512, 0, stream>>>(hbf, wqg, wqu, wqd, counts, list,
                                              ybuf);
  moe_combine_kernel<<<kT, 256, 0, stream>>>(ybuf, topw, out);
}
